// ComplexConv4dNet_53626961658171
// MI455X (gfx1250) — compile-verified, matrix operand order checked by reading, not run on hardware
//
#include <hip/hip_runtime.h>
#include <hip/hip_bf16.h>

typedef __attribute__((ext_vector_type(16))) _Float16 v16h;
typedef __attribute__((ext_vector_type(8)))  _Float16 v8h;
typedef __attribute__((ext_vector_type(8)))  float    v8f;

#define S_TOT 41472   // 2 * 12^4
#define VOX   20736   // 12^4

__device__ __forceinline__ void decomp(int s, int& n, int& a, int& b, int& c, int& d) {
    n = s / VOX;  int r = s - n * VOX;
    a = r / 1728; r -= a * 1728;
    b = r / 144;  r -= b * 144;
    c = r / 12;   d  = r - c * 12;
}

__device__ __forceinline__ void tapoff(int t, int& da, int& db, int& dc, int& dd) {
    da = t / 27 - 1; int r = t % 27;
    db = r / 9  - 1; r %= 9;
    dc = r / 3  - 1; dd = r % 3 - 1;
}

// ---------------------------------------------------------------------------
// Repack conv weights (Cout, Cin, 81) f32 -> B-fragment-ordered f16:
//   bp[((t*KCH + kc)*NT + nt)*32 + lane][j] with
//   K(j) = (j<8 ? j : j+8) + (lane>=16 ? 8 : 0), ci = kc*32 + K(j),
//   co = nt*16 + (lane&15)   (mirror of the ISA 16-bit A-matrix layout).
// ---------------------------------------------------------------------------
__global__ void pack_b_kernel(const float* __restrict__ w, _Float16* __restrict__ bp,
                              int CIN, int COUT) {
    int idx = blockIdx.x * blockDim.x + threadIdx.x;
    int KCH = CIN >> 5, NT = COUT >> 4;
    int total = 81 * KCH * NT * 32;
    if (idx >= total) return;
    int lane = idx & 31; int rest = idx >> 5;
    int nt = rest % NT; rest /= NT;
    int kc = rest % KCH; int t = rest / KCH;
    int co    = nt * 16 + (lane & 15);
    int khalf = (lane >> 4) << 3;
    _Float16* dst = bp + (long)idx * 16;
#pragma unroll
    for (int j = 0; j < 16; ++j) {
        int K  = (j < 8 ? j : j + 8) + khalf;
        int ci = kc * 32 + K;
        dst[j] = (_Float16)w[((long)co * CIN + ci) * 81 + t];
    }
}

// ---------------------------------------------------------------------------
// Layer 1: Cin=1 direct conv (K=81). One spatial point per block; 81 taps of
// the input staged in LDS, 64 threads each produce one output channel.
// ---------------------------------------------------------------------------
__global__ void conv1_kernel(const float* __restrict__ x, const float* __restrict__ w1,
                             float* __restrict__ out) {
    __shared__ float xt[81];
    int s = blockIdx.x;
    int n, a, b, c, d; decomp(s, n, a, b, c, d);
    int tid = threadIdx.x;
    if (tid < 81) {
        int da, db, dc, dd; tapoff(tid, da, db, dc, dd);
        int aa = a + da, bb = b + db, cc = c + dc, de = d + dd;
        bool valid = ((unsigned)aa < 12u) & ((unsigned)bb < 12u) &
                     ((unsigned)cc < 12u) & ((unsigned)de < 12u);
        xt[tid] = valid ? x[n * VOX + aa * 1728 + bb * 144 + cc * 12 + de] : 0.f;
    }
    __syncthreads();
    if (tid < 64) {
        float sum = 0.f;
#pragma unroll 9
        for (int t = 0; t < 81; ++t) sum += xt[t] * w1[tid * 81 + t];
        out[(long)s * 64 + tid] = sum;   // conv bias cancels in training-mode BN
    }
}

// ---------------------------------------------------------------------------
// Layers 2/3: shifted implicit GEMM on the WMMA pipe.
// One wave computes a 16x32 output slab (two N-tiles): each A fragment feeds
// two v_wmma_f32_16x16x32_f16 issues. Tap deltas advance as a base-3 odometer
// (wave-uniform SALU) with the linear offset recomputed per tap, so per-tap VALU is
// just 4 adds + 4 range checks per lane. Boundary taps zero-fill A with EXEC
// re-converged before the WMMAs.
// ---------------------------------------------------------------------------
template <int CIN, int COUT>
__global__ void __launch_bounds__(32)
conv_wmma_kernel(const _Float16* __restrict__ act,   // [S][CIN] f16
                 const _Float16* __restrict__ bp,    // packed fragments
                 float* __restrict__ out) {          // [S][COUT] f32
    constexpr int KCH = CIN >> 5;
    constexpr int NT  = COUT >> 4;
    const int lane = threadIdx.x;
    const int mt   = blockIdx.x;
    const int nt0  = blockIdx.y * 2;     // first of two N-tiles
    const int m    = lane & 15;
    const int s    = mt * 16 + m;
    int n, a, b, c, d; decomp(s, n, a, b, c, d);
    const int khalf = (lane >> 4) << 3;  // 0 for lanes 0-15, 8 for lanes 16-31

    v8f acc0 = {}, acc1 = {};
    int da = -1, db = -1, dc = -1, dd = -1;          // odometer (uniform)
    for (int t = 0; t < 81; ++t) {
        int aa = a + da, bb = b + db, cc = c + dc, de = d + dd;
        bool valid = ((unsigned)aa < 12u) & ((unsigned)bb < 12u) &
                     ((unsigned)cc < 12u) & ((unsigned)de < 12u);
        int dlin = da * 1728 + db * 144 + dc * 12 + dd;   // uniform
        const _Float16* arow = act + (long)(s + dlin) * CIN;
        const _Float16* btap = bp + ((long)(t * KCH) * NT) * 512 + lane * 16;
        __builtin_prefetch(btap + (long)KCH * NT * 512, 0, 1);  // next tap weights
#pragma unroll
        for (int kc = 0; kc < KCH; ++kc) {
            v8h lo = {}, hi = {};
            if (valid) {
                const v8h* p = (const v8h*)(arow + kc * 32 + khalf);
                lo = p[0];   // channels [c0 .. c0+7]
                hi = p[2];   // channels [c0+16 .. c0+23]
            }
            union { v16h v; v8h h[2]; } af;
            af.h[0] = lo; af.h[1] = hi;
            v16h bf0 = *(const v16h*)(btap + ((long)kc * NT + nt0) * 512);
            v16h bf1 = *(const v16h*)(btap + ((long)kc * NT + nt0 + 1) * 512);
            acc0 = __builtin_amdgcn_wmma_f32_16x16x32_f16(
                false, af.v, false, bf0, (short)0, acc0, false, false);
            acc1 = __builtin_amdgcn_wmma_f32_16x16x32_f16(
                false, af.v, false, bf1, (short)0, acc1, false, false);
        }
        // advance base-3 odometer over (da,db,dc,dd)
        if (++dd == 2) { dd = -1;
            if (++dc == 2) { dc = -1;
                if (++db == 2) { db = -1; ++da; } } }
    }
    // D layout: VGPR r -> row (r + (lane>=16 ? 8 : 0)), col = lane&15
    const int co0 = nt0 * 16 + (lane & 15);
    const int mb  = mt * 16 + ((lane >> 4) << 3);
#pragma unroll
    for (int r = 0; r < 8; ++r) {
        out[(long)(mb + r) * COUT + co0]      = acc0[r];
        out[(long)(mb + r) * COUT + co0 + 16] = acc1[r];
    }
}

// ---------------------------------------------------------------------------
// BatchNorm: per-channel sum / sumsq via LDS float atomics, then scale/shift,
// then fused normalize+ReLU writing f16 activations (channels-last).
// ---------------------------------------------------------------------------
__global__ void zero_stats_kernel(float* __restrict__ stats) {
    stats[threadIdx.x] = 0.f;   // 256 floats: sums[128] + sumsq[128]
}

__global__ void bn_stats_kernel(const float* __restrict__ raw,
                                float* __restrict__ sums, float* __restrict__ sumsq,
                                int C, long total) {
    __shared__ float lsum[128], lsq[128];
    int tid = threadIdx.x;
    for (int i = tid; i < C; i += blockDim.x) { lsum[i] = 0.f; lsq[i] = 0.f; }
    __syncthreads();
    const int cmask = C - 1;
    for (long i = (long)blockIdx.x * blockDim.x + tid; i < total;
         i += (long)gridDim.x * blockDim.x) {
        float v = raw[i];
        int ch = (int)(i & cmask);
        atomicAdd(&lsum[ch], v);       // ds_add_f32
        atomicAdd(&lsq[ch], v * v);
    }
    __syncthreads();
    for (int i = tid; i < C; i += blockDim.x) {
        atomicAdd(&sums[i], lsum[i]);
        atomicAdd(&sumsq[i], lsq[i]);
    }
}

__global__ void bn_finalize_kernel(const float* __restrict__ sums,
                                   const float* __restrict__ sumsq,
                                   const float* __restrict__ g,
                                   const float* __restrict__ be,
                                   float* __restrict__ scale, float* __restrict__ shift,
                                   int C) {
    int c = blockIdx.x * blockDim.x + threadIdx.x;
    if (c >= C) return;
    float inv_n = 1.f / (float)S_TOT;
    float mean = sums[c] * inv_n;
    float var  = sumsq[c] * inv_n - mean * mean;
    float sc   = g[c] * rsqrtf(var + 1e-5f);
    scale[c] = sc;
    shift[c] = be[c] - mean * sc;
}

__global__ void bn_apply_kernel(const float* __restrict__ raw,
                                const float* __restrict__ scale,
                                const float* __restrict__ shift,
                                _Float16* __restrict__ act, int C, long total) {
    const int cmask = C - 1;
    for (long i = (long)blockIdx.x * blockDim.x + threadIdx.x; i < total;
         i += (long)gridDim.x * blockDim.x) {
        int ch = (int)(i & cmask);
        float v = fmaf(raw[i], scale[ch], shift[ch]);
        act[i] = (_Float16)fmaxf(v, 0.f);
    }
}

// ---------------------------------------------------------------------------
// Layer 4: Cout=1, dot of 81 taps x 64 channels per voxel, + bias, sigmoid.
// ---------------------------------------------------------------------------
__global__ void conv4_kernel(const _Float16* __restrict__ act,   // [S][64]
                             const float* __restrict__ w4,       // [64][81]
                             const float* __restrict__ b4,
                             float* __restrict__ out) {
    int s = blockIdx.x * blockDim.x + threadIdx.x;
    if (s >= S_TOT) return;
    int n, a, b, c, d; decomp(s, n, a, b, c, d);
    float sum = b4[0];
    int da = -1, db = -1, dc = -1, dd = -1;
    for (int t = 0; t < 81; ++t) {
        int aa = a + da, bb = b + db, cc = c + dc, de = d + dd;
        bool valid = ((unsigned)aa < 12u) & ((unsigned)bb < 12u) &
                     ((unsigned)cc < 12u) & ((unsigned)de < 12u);
        int dlin = da * 1728 + db * 144 + dc * 12 + dd;
        if (++dd == 2) { dd = -1;
            if (++dc == 2) { dc = -1;
                if (++db == 2) { db = -1; ++da; } } }
        if (!valid) continue;
        const _Float16* row = act + (long)(s + dlin) * 64;
#pragma unroll 8
        for (int ci = 0; ci < 64; ++ci)
            sum += (float)row[ci] * w4[ci * 81 + t];
    }
    out[s] = 1.f / (1.f + expf(-sum));
}

// ---------------------------------------------------------------------------
extern "C" void kernel_launch(void* const* d_in, const int* in_sizes, int n_in,
                              void* d_out, int out_size, void* d_ws, size_t ws_size,
                              hipStream_t stream) {
    (void)in_sizes; (void)n_in; (void)out_size; (void)ws_size;
    const float* x   = (const float*)d_in[0];
    const float* w1  = (const float*)d_in[1];
    const float* g1  = (const float*)d_in[3];
    const float* be1 = (const float*)d_in[4];
    const float* w2  = (const float*)d_in[5];
    const float* g2  = (const float*)d_in[7];
    const float* be2 = (const float*)d_in[8];
    const float* w3  = (const float*)d_in[9];
    const float* g3  = (const float*)d_in[11];
    const float* be3 = (const float*)d_in[12];
    const float* w4  = (const float*)d_in[13];
    const float* b4  = (const float*)d_in[14];
    float* out = (float*)d_out;

    // Workspace carve-up (all regions 256B aligned; total ~40 MB)
    char* ws = (char*)d_ws;
    size_t o = 0;
    float*    raw  = (float*)(ws + o);     o += (size_t)S_TOT * 128 * 4; // conv raw f32
    _Float16* actA = (_Float16*)(ws + o);  o += (size_t)S_TOT * 64 * 2;  // 64-ch f16
    _Float16* actB = (_Float16*)(ws + o);  o += (size_t)S_TOT * 128 * 2; // 128-ch f16
    _Float16* b2p  = (_Float16*)(ws + o);  o += (size_t)81 * 2 * 8 * 512 * 2;
    _Float16* b3p  = (_Float16*)(ws + o);  o += (size_t)81 * 4 * 4 * 512 * 2;
    float*    stats = (float*)(ws + o);    // [0:128)=sum [128:256)=sumsq
                                           // [256:384)=scale [384:512)=shift

    // Weight repack (deterministic, every call)
    pack_b_kernel<<<162, 256, 0, stream>>>(w2, b2p, 64, 128);
    pack_b_kernel<<<162, 256, 0, stream>>>(w3, b3p, 128, 64);

    // ---- Layer 1: conv(1->64) + BN + ReLU ----
    conv1_kernel<<<S_TOT, 96, 0, stream>>>(x, w1, raw);
    zero_stats_kernel<<<1, 256, 0, stream>>>(stats);
    bn_stats_kernel<<<512, 256, 0, stream>>>(raw, stats, stats + 128, 64, (long)S_TOT * 64);
    bn_finalize_kernel<<<1, 128, 0, stream>>>(stats, stats + 128, g1, be1,
                                              stats + 256, stats + 384, 64);
    bn_apply_kernel<<<1024, 256, 0, stream>>>(raw, stats + 256, stats + 384,
                                              actA, 64, (long)S_TOT * 64);

    // ---- Layer 2: conv(64->128) via WMMA + BN + ReLU ----
    conv_wmma_kernel<64, 128><<<dim3(S_TOT / 16, 4), 32, 0, stream>>>(actA, b2p, raw);
    zero_stats_kernel<<<1, 256, 0, stream>>>(stats);
    bn_stats_kernel<<<512, 256, 0, stream>>>(raw, stats, stats + 128, 128, (long)S_TOT * 128);
    bn_finalize_kernel<<<1, 128, 0, stream>>>(stats, stats + 128, g2, be2,
                                              stats + 256, stats + 384, 128);
    bn_apply_kernel<<<1024, 256, 0, stream>>>(raw, stats + 256, stats + 384,
                                              actB, 128, (long)S_TOT * 128);

    // ---- Layer 3: conv(128->64) via WMMA + BN + ReLU ----
    conv_wmma_kernel<128, 64><<<dim3(S_TOT / 16, 2), 32, 0, stream>>>(actB, b3p, raw);
    zero_stats_kernel<<<1, 256, 0, stream>>>(stats);
    bn_stats_kernel<<<512, 256, 0, stream>>>(raw, stats, stats + 128, 64, (long)S_TOT * 64);
    bn_finalize_kernel<<<1, 128, 0, stream>>>(stats, stats + 128, g3, be3,
                                              stats + 256, stats + 384, 64);
    bn_apply_kernel<<<1024, 256, 0, stream>>>(raw, stats + 256, stats + 384,
                                              actA, 64, (long)S_TOT * 64);

    // ---- Layer 4: conv(64->1) + bias + sigmoid ----
    conv4_kernel<<<(S_TOT + 127) / 128, 128, 0, stream>>>(actA, w4, b4, out);
}
